// Att_87411174408394
// MI455X (gfx1250) — compile-verified, matrix operand order checked by reading, not run on hardware
//
#include <hip/hip_runtime.h>

typedef _Float16 v16h __attribute__((ext_vector_type(16)));
typedef _Float16 v8h  __attribute__((ext_vector_type(8)));
typedef float    v8f  __attribute__((ext_vector_type(8)));
typedef float    v4f  __attribute__((ext_vector_type(4)));

constexpr int DM = 128;   // feature dim
constexpr int TM = 16;    // rows (edges/agents) per block

// ---------------------------------------------------------------------------
// Wave-level 16xK * Kx16 -> 16x16 f32 tile. A: row-major f16 [16][lda] (LDS),
// B: Wt row-major f16 [128][K] (global, n-major = transposed weight).
// Fragment layouts follow CDNA5 ISA 7.12.2 (16-bit A 16x32, B symmetric).
// ---------------------------------------------------------------------------
__device__ __forceinline__ v8f wmma_acc_16xK(const _Float16* __restrict__ A, int lda, int K,
                                             const _Float16* __restrict__ Wt, int tid) {
  const int lane = tid & 31;
  const int wave = tid >> 5;            // n-tile 0..7
  const int kh   = (lane >> 4) << 3;    // 0 or 8 halves (K-half select)
  const _Float16* pa = A  + (lane & 15) * lda + kh;
  const _Float16* pb = Wt + (size_t)(wave * 16 + (lane & 15)) * K + kh;
  v8f acc = {};
  const int nk = K >> 5;
  for (int kc = 0; kc < nk; ++kc) {
    v8h alo = *(const v8h*)(pa);
    v8h ahi = *(const v8h*)(pa + 16);
    v8h blo = *(const v8h*)(pb);
    v8h bhi = *(const v8h*)(pb + 16);
    v16h a, b;
#pragma unroll
    for (int i = 0; i < 8; ++i) { a[i] = alo[i]; a[i + 8] = ahi[i]; b[i] = blo[i]; b[i + 8] = bhi[i]; }
    acc = __builtin_amdgcn_wmma_f32_16x16x32_f16(false, a, false, b, (short)0, acc, false, false);
    pa += 32; pb += 32;
  }
  return acc;
}

// D-tile (VGPR r: lanes0-15 M=r, lanes16-31 M=8+r; N=lane&15) -> LDS [16][128]
__device__ __forceinline__ void store_tile_lds(float* sOut, const v8f& acc, int tid) {
  const int lane = tid & 31;
  const int colg = (tid >> 5) * 16 + (lane & 15);
  const int rb   = (lane >> 4) << 3;
#pragma unroll
  for (int r = 0; r < 8; ++r) sOut[(rb + r) * DM + colg] = acc[r];
}

// Convert 8 f32 (two float4) -> v8h
__device__ __forceinline__ v8h cvt8(const v4f& a, const v4f& b) {
  v8h h;
#pragma unroll
  for (int i = 0; i < 4; ++i) { h[i] = (_Float16)a[i]; h[4 + i] = (_Float16)b[i]; }
  return h;
}

// Gather one 128-f32 row -> f16 LDS row (thread handles 8 contiguous cols)
__device__ __forceinline__ void gather_row_f16(const float* __restrict__ src, size_t srow,
                                               _Float16* dstRow, int c0) {
  const v4f* p = (const v4f*)(src + srow * DM) + 2 * c0;
  v4f a = p[0], b = p[1];
  *(v8h*)(dstRow + 8 * c0) = cvt8(a, b);
}

// Per-row layernorm stats over 128 channels (16 threads/row, contiguous 8-chunks,
// 16-lane xor-butterfly reduction -- wave32-safe, masks stay in 16-lane half).
__device__ __forceinline__ void gn_row(const float* sS, int tid, v4f& x0, v4f& x1,
                                       float& mu, float& rstd) {
  const int row = tid >> 4, c0 = tid & 15;
  const v4f* p = (const v4f*)(sS + row * DM) + 2 * c0;
  x0 = p[0]; x1 = p[1];
  float s = 0.f, s2 = 0.f;
#pragma unroll
  for (int i = 0; i < 4; ++i) { s += x0[i] + x1[i]; s2 += x0[i] * x0[i] + x1[i] * x1[i]; }
#pragma unroll
  for (int m = 1; m < 16; m <<= 1) { s += __shfl_xor(s, m, 32); s2 += __shfl_xor(s2, m, 32); }
  mu = s * (1.f / 128.f);
  rstd = rsqrtf(s2 * (1.f / 128.f) - mu * mu + 1e-5f);
}

// GN + ReLU + f16 convert + vectorized LDS store (dst has row stride ldd)
__device__ __forceinline__ void gn_store_f16(const float* sS,
                                             const float* __restrict__ g,
                                             const float* __restrict__ b,
                                             _Float16* dst, int ldd, int tid) {
  v4f x0, x1; float mu, rstd;
  gn_row(sS, tid, x0, x1, mu, rstd);
  const int row = tid >> 4, c0 = tid & 15;
  v4f g0 = ((const v4f*)g)[2 * c0], g1 = ((const v4f*)g)[2 * c0 + 1];
  v4f b0 = ((const v4f*)b)[2 * c0], b1 = ((const v4f*)b)[2 * c0 + 1];
  v8h h;
#pragma unroll
  for (int i = 0; i < 4; ++i) {
    h[i]     = (_Float16)fmaxf((x0[i] - mu) * rstd * g0[i] + b0[i], 0.f);
    h[4 + i] = (_Float16)fmaxf((x1[i] - mu) * rstd * g1[i] + b1[i], 0.f);
  }
  *(v8h*)(dst + row * ldd + 8 * c0) = h;
}

// ---------------------------------------------------------------------------
// Weight prep: dst[n*K+k] = (f16) src[k*N+n]   (src is [K_in][N_out] row-major)
// ---------------------------------------------------------------------------
__global__ void convert_transpose_kernel(const float* __restrict__ src,
                                         _Float16* __restrict__ dst, int K, int N) {
  long i = (long)blockIdx.x * blockDim.x + threadIdx.x;
  if (i >= (long)K * N) return;
  int n = (int)(i / K);
  int k = (int)(i % K);
  dst[i] = (_Float16)src[(size_t)k * N + n];
}

// ---------------------------------------------------------------------------
// a_buf = agts @ W_agt   (16 agents per block; D routed via LDS -> b128 stores)
// ---------------------------------------------------------------------------
__global__ __launch_bounds__(256) void agent_gemm_kernel(const float* __restrict__ agts,
                                                         const _Float16* __restrict__ WtAgt,
                                                         float* __restrict__ a_buf, int N) {
  __shared__ __align__(16) _Float16 sA[TM * DM];
  __shared__ __align__(16) float    sS[TM * DM];
  const int tid = threadIdx.x;
  const long r0 = (long)blockIdx.x * TM;
  const int row = tid >> 4, c0 = tid & 15;
  long gr = r0 + row; if (gr >= N) gr = N - 1;
  gather_row_f16(agts, (size_t)gr, sA + row * DM, c0);
  __syncthreads();
  v8f acc = wmma_acc_16xK(sA, DM, DM, WtAgt, tid);
  store_tile_lds(sS, acc, tid);
  __syncthreads();
  if (r0 + row < N) {
    const v4f* p = (const v4f*)(sS + row * DM) + 2 * c0;
    v4f x0 = p[0], x1 = p[1];
    v4f* dst = (v4f*)(a_buf + (size_t)(r0 + row) * DM) + 2 * c0;
    dst[0] = x0; dst[1] = x1;
  }
}

// ---------------------------------------------------------------------------
// Fused per-edge pipeline. Both gathers (ctx[wi], agts[hi]) issue up-front so
// their HBM latency overlaps the dist-MLP and first GEMM. 16 edges / block.
// ---------------------------------------------------------------------------
__global__ __launch_bounds__(256) void edge_kernel(
    const float* __restrict__ agts, const float* __restrict__ ctx,
    const float* __restrict__ agt_ctrs, const float* __restrict__ ctx_ctrs,
    const float* __restrict__ W_dist1, const float* __restrict__ b_dist1,
    const float* __restrict__ g_dist, const float* __restrict__ b_dist,
    const float* __restrict__ g_q, const float* __restrict__ b_q,
    const float* __restrict__ g_c1, const float* __restrict__ b_c1,
    const int* __restrict__ hi, const int* __restrict__ wi,
    const _Float16* __restrict__ WtD2, const _Float16* __restrict__ WtQ,
    const _Float16* __restrict__ WtC1, const _Float16* __restrict__ WtC2,
    float* __restrict__ a_buf, int E) {
  __shared__ __align__(16) _Float16 sA[TM * DM];          // d1, later c1
  __shared__ __align__(16) _Float16 sAgt[TM * DM];        // gathered agts[hi]
  __shared__ __align__(16) _Float16 sCat[TM * 3 * DM];    // [d2 | q | ctx] f16
  __shared__ __align__(16) float    sS[TM * DM];          // f32 D tile
  __shared__ int sHi[TM], sWi[TM];
  __shared__ float sDx[TM], sDy[TM];

  const int tid = threadIdx.x;
  const long e0 = (long)blockIdx.x * TM;

  if (tid < TM) {
    long e = e0 + tid;
    int h = 0, w = 0;
    if (e < E) { h = hi[e]; w = wi[e]; }
    sHi[tid] = h; sWi[tid] = w;
    sDx[tid] = agt_ctrs[2 * h]     - ctx_ctrs[2 * w];
    sDy[tid] = agt_ctrs[2 * h + 1] - ctx_ctrs[2 * w + 1];
  }
  __syncthreads();

  const int row = tid >> 4, c0 = tid & 15;
  const int h = sHi[row], w = sWi[row];

  // Issue both gathers first (independent b128 loads -> MLP/GEMM overlap).
  gather_row_f16(ctx,  (size_t)w, sCat + row * 3 * DM + 2 * DM, c0);
  gather_row_f16(agts, (size_t)h, sAgt + row * DM, c0);

  // d1 = relu(ctr_diff @ W_dist1 + b)  (vectorized over 8 contiguous cols)
  {
    const float dx = sDx[row], dy = sDy[row];
    v4f w00 = ((const v4f*)W_dist1)[2 * c0],        w01 = ((const v4f*)W_dist1)[2 * c0 + 1];
    v4f w10 = ((const v4f*)(W_dist1 + DM))[2 * c0], w11 = ((const v4f*)(W_dist1 + DM))[2 * c0 + 1];
    v4f bb0 = ((const v4f*)b_dist1)[2 * c0],        bb1 = ((const v4f*)b_dist1)[2 * c0 + 1];
    v8h hh;
#pragma unroll
    for (int i = 0; i < 4; ++i) {
      hh[i]     = (_Float16)fmaxf(fmaf(dx, w00[i], fmaf(dy, w10[i], bb0[i])), 0.f);
      hh[4 + i] = (_Float16)fmaxf(fmaf(dx, w01[i], fmaf(dy, w11[i], bb1[i])), 0.f);
    }
    *(v8h*)(sA + row * DM + 8 * c0) = hh;
  }
  __syncthreads();

  // d2 = relu(gn(d1 @ W_dist2)) -> concat seg0
  { v8f acc = wmma_acc_16xK(sA, DM, DM, WtD2, tid); store_tile_lds(sS, acc, tid); }
  __syncthreads();
  gn_store_f16(sS, g_dist, b_dist, sCat, 3 * DM, tid);
  __syncthreads();

  // q = relu(gn(agts[hi] @ W_q)) -> concat seg1
  { v8f acc = wmma_acc_16xK(sAgt, DM, DM, WtQ, tid); store_tile_lds(sS, acc, tid); }
  __syncthreads();
  gn_store_f16(sS, g_q, b_q, sCat + DM, 3 * DM, tid);
  __syncthreads();

  // c1 = relu(gn(cat @ W_c1))  (K = 384) -> sA
  { v8f acc = wmma_acc_16xK(sCat, 3 * DM, 3 * DM, WtC1, tid); store_tile_lds(sS, acc, tid); }
  __syncthreads();
  gn_store_f16(sS, g_c1, b_c1, sA, DM, tid);
  __syncthreads();

  // c2 = c1 @ W_c2 -> atomic scatter into a_buf[hi]
  { v8f acc = wmma_acc_16xK(sA, DM, DM, WtC2, tid); store_tile_lds(sS, acc, tid); }
  __syncthreads();

  if (e0 + row < E) {
    const v4f* p = (const v4f*)(sS + row * DM) + 2 * c0;
    v4f x0 = p[0], x1 = p[1];
    float* dst = a_buf + (size_t)h * DM + 8 * c0;
#pragma unroll
    for (int i = 0; i < 4; ++i) { atomicAdd(dst + i, x0[i]); atomicAdd(dst + 4 + i, x1[i]); }
  }
}

// ---------------------------------------------------------------------------
// out = relu(gn(relu(gn(a_buf)) @ W_lin) + agts)
// ---------------------------------------------------------------------------
__global__ __launch_bounds__(256) void post_kernel(
    const float* __restrict__ agts, const float* __restrict__ a_buf,
    const float* __restrict__ g_n, const float* __restrict__ b_n,
    const _Float16* __restrict__ WtLin,
    const float* __restrict__ g_lin, const float* __restrict__ b_lin,
    float* __restrict__ out, int N) {
  __shared__ __align__(16) _Float16 sA[TM * DM];
  __shared__ __align__(16) float    sS[TM * DM];
  const int tid = threadIdx.x;
  const long r0 = (long)blockIdx.x * TM;
  const int row = tid >> 4, c0 = tid & 15;
  long gr = r0 + row; if (gr >= N) gr = N - 1;
  {
    const v4f* p = (const v4f*)(a_buf + (size_t)gr * DM) + 2 * c0;
    v4f x0 = p[0], x1 = p[1];
    v4f* q = (v4f*)(sS + row * DM) + 2 * c0;
    q[0] = x0; q[1] = x1;
  }
  __syncthreads();
  gn_store_f16(sS, g_n, b_n, sA, DM, tid);   // relu(gn(.)) -> f16 A
  __syncthreads();
  { v8f acc = wmma_acc_16xK(sA, DM, DM, WtLin, tid); store_tile_lds(sS, acc, tid); }
  __syncthreads();
  {
    v4f x0, x1; float mu, rstd;
    gn_row(sS, tid, x0, x1, mu, rstd);
    if (r0 + row < N) {
      v4f g0 = ((const v4f*)g_lin)[2 * c0], g1 = ((const v4f*)g_lin)[2 * c0 + 1];
      v4f b0 = ((const v4f*)b_lin)[2 * c0], b1 = ((const v4f*)b_lin)[2 * c0 + 1];
      const v4f* rp = (const v4f*)(agts + (size_t)(r0 + row) * DM) + 2 * c0;
      v4f ra = rp[0], rb = rp[1];
      v4f o0, o1;
#pragma unroll
      for (int i = 0; i < 4; ++i) {
        o0[i] = fmaxf((x0[i] - mu) * rstd * g0[i] + b0[i] + ra[i], 0.f);
        o1[i] = fmaxf((x1[i] - mu) * rstd * g1[i] + b1[i] + rb[i], 0.f);
      }
      v4f* op = (v4f*)(out + (size_t)(r0 + row) * DM) + 2 * c0;
      op[0] = o0; op[1] = o1;
    }
  }
}

// ---------------------------------------------------------------------------
extern "C" void kernel_launch(void* const* d_in, const int* in_sizes, int n_in,
                              void* d_out, int out_size, void* d_ws, size_t ws_size,
                              hipStream_t stream) {
  const float* agts     = (const float*)d_in[0];
  const float* ctx      = (const float*)d_in[1];
  const float* agt_ctrs = (const float*)d_in[2];
  const float* ctx_ctrs = (const float*)d_in[3];
  const float* W_dist1  = (const float*)d_in[4];
  const float* b_dist1  = (const float*)d_in[5];
  const float* W_dist2  = (const float*)d_in[6];
  const float* g_dist   = (const float*)d_in[7];
  const float* b_dist   = (const float*)d_in[8];
  const float* W_q      = (const float*)d_in[9];
  const float* g_q      = (const float*)d_in[10];
  const float* b_q      = (const float*)d_in[11];
  const float* W_c1     = (const float*)d_in[12];
  const float* g_c1     = (const float*)d_in[13];
  const float* b_c1     = (const float*)d_in[14];
  const float* W_c2     = (const float*)d_in[15];
  const float* W_agt    = (const float*)d_in[16];
  const float* g_n      = (const float*)d_in[17];
  const float* b_n      = (const float*)d_in[18];
  const float* W_lin    = (const float*)d_in[19];
  const float* g_lin    = (const float*)d_in[20];
  const float* b_lin    = (const float*)d_in[21];
  const int*   hi       = (const int*)d_in[22];
  const int*   wi       = (const int*)d_in[23];

  const int N_AGT = in_sizes[2] / 2;
  const int E     = in_sizes[22];

  // Workspace layout: a_buf (f32 [N_AGT][128]) then f16 transposed weights.
  char* ws = (char*)d_ws;
  float* a_buf = (float*)ws;
  size_t off = ((size_t)N_AGT * DM * sizeof(float) + 255) & ~(size_t)255;
  _Float16* WtD2  = (_Float16*)(ws + off); off += (size_t)DM * DM * sizeof(_Float16);
  _Float16* WtQ   = (_Float16*)(ws + off); off += (size_t)DM * DM * sizeof(_Float16);
  _Float16* WtC1  = (_Float16*)(ws + off); off += (size_t)DM * 3 * DM * sizeof(_Float16);
  _Float16* WtC2  = (_Float16*)(ws + off); off += (size_t)DM * DM * sizeof(_Float16);
  _Float16* WtAgt = (_Float16*)(ws + off); off += (size_t)DM * DM * sizeof(_Float16);
  _Float16* WtLin = (_Float16*)(ws + off); off += (size_t)DM * DM * sizeof(_Float16);
  (void)n_in; (void)out_size; (void)ws_size;

  const int sq  = DM * DM;        // 16384
  const int sq3 = DM * 3 * DM;    // 49152
  convert_transpose_kernel<<<(sq  + 255) / 256, 256, 0, stream>>>(W_dist2, WtD2, DM, DM);
  convert_transpose_kernel<<<(sq  + 255) / 256, 256, 0, stream>>>(W_q,     WtQ,  DM, DM);
  convert_transpose_kernel<<<(sq3 + 255) / 256, 256, 0, stream>>>(W_c1,    WtC1, 3 * DM, DM);
  convert_transpose_kernel<<<(sq  + 255) / 256, 256, 0, stream>>>(W_c2,    WtC2, DM, DM);
  convert_transpose_kernel<<<(sq  + 255) / 256, 256, 0, stream>>>(W_agt,   WtAgt, DM, DM);
  convert_transpose_kernel<<<(sq  + 255) / 256, 256, 0, stream>>>(W_lin,   WtLin, DM, DM);

  agent_gemm_kernel<<<(N_AGT + TM - 1) / TM, 256, 0, stream>>>(agts, WtAgt, a_buf, N_AGT);

  edge_kernel<<<(E + TM - 1) / TM, 256, 0, stream>>>(
      agts, ctx, agt_ctrs, ctx_ctrs, W_dist1, b_dist1,
      g_dist, b_dist, g_q, b_q, g_c1, b_c1, hi, wi,
      WtD2, WtQ, WtC1, WtC2, a_buf, E);

  post_kernel<<<(N_AGT + TM - 1) / TM, 256, 0, stream>>>(
      agts, a_buf, g_n, b_n, WtLin, g_lin, b_lin, (float*)d_out, N_AGT);
}
